// EnsembleCritic_79448305041731
// MI455X (gfx1250) — compile-verified, matrix operand order checked by reading, not run on hardware
//
#include <hip/hip_runtime.h>
#include <hip/hip_bf16.h>

// ---------------------------------------------------------------------------
// EnsembleCritic fused forward for gfx1250 (MI455X).
// Pass 1: convert weights fp32->bf16 into d_ws, pre-transposed into the exact
//         padded per-chunk LDS fragment image (1024 rows x 40 elems / chunk);
//         convert x to bf16.
// Pass 2: one workgroup = (ensemble e, 64-row batch tile); carries the tile
//         through GEMM1 -> LN+ReLU -> GEMM2 -> LN+ReLU -> head entirely in
//         LDS/VGPRs with v_wmma_f32_16x16x32_bf16. Weight chunks stream from
//         L2 as bf16 into double-buffered LDS via async global->LDS DMA
//         (ASYNCcnt); one barrier per K-chunk, staging overlaps WMMA.
// Pass 3: q[b] = min_e qs[e][b].
// ---------------------------------------------------------------------------

#define B_TOT    16384
#define D_IN_K   512
#define H_N      1024
#define E_N      8
#define BT       64        // batch rows per workgroup
#define NTHREADS 512       // 16 waves (wave32)
#define LDB      40        // padded K-stride (elems): 80B rows, 16B-aligned, conflict-free
#define LDH      1032      // stride (elems) of the 64 x 1024 h1 LDS tile
#define SB_ELEMS (H_N * LDB)   // 40960 elems = 81920 B per weight chunk image
#define SA_ELEMS (BT * LDB)    // 2560 elems  = 5120 B per A buffer
#define NCHUNK1  (D_IN_K / 32) // 16 chunks for W1 per ensemble
#define NCHUNK2  (H_N / 32)    // 32 chunks for W2 per ensemble
#define WS_W2OFF (E_N * NCHUNK1)   // chunk index base of W2 images
#define LN_EPS   1e-5f

typedef __bf16 bf16_t;
typedef __attribute__((ext_vector_type(16))) __bf16 v16bf;
typedef __attribute__((ext_vector_type(8)))  __bf16 v8bf;
typedef __attribute__((ext_vector_type(4)))  __bf16 v4bf;
typedef __attribute__((ext_vector_type(2)))  __bf16 v2bf;
typedef __attribute__((ext_vector_type(8)))  float  v8f;
typedef __attribute__((ext_vector_type(4)))  int    v4i;

union FragU { v16bf v; v8bf h[2]; };

// One 16x32 bf16 WMMA operand fragment (ISA 16-bit A layout):
// lanes 0-15: K={0..7,16..23}, lanes 16-31: K={8..15,24..31}; row = lane&15.
__device__ __forceinline__ v16bf load_frag(const bf16_t* row, int hi) {
  FragU u;
  u.h[0] = *reinterpret_cast<const v8bf*>(row + (hi ? 8 : 0));
  u.h[1] = *reinterpret_cast<const v8bf*>(row + (hi ? 24 : 16));
  return u.v;
}

// Stage one 32(K) x 1024(N) fp32 chunk -> bf16, K-transposed (sB[n][k]).
// (Used only by the one-time conversion kernel.)
__device__ __forceinline__ void stage_w_chunk(const float* __restrict__ Wl,
                                              int kc, bf16_t* sB, int tid) {
  float4 ca = {}, cb = {}, na = {}, nb = {};
  {
    int kp = tid >> 8, n4 = (tid & 255) << 2;
    const float* p = Wl + (size_t)(kc + (kp << 1)) * H_N + n4;
    ca = *reinterpret_cast<const float4*>(p);
    cb = *reinterpret_cast<const float4*>(p + H_N);
  }
#pragma unroll
  for (int u = 0; u < 8; ++u) {
    if (u < 7) {
      int lin = tid + (u + 1) * NTHREADS;
      int kp = lin >> 8, n4 = (lin & 255) << 2;
      const float* p = Wl + (size_t)(kc + (kp << 1)) * H_N + n4;
      na = *reinterpret_cast<const float4*>(p);
      nb = *reinterpret_cast<const float4*>(p + H_N);
    }
    {
      int lin = tid + u * NTHREADS;
      int kp = lin >> 8, n4 = (lin & 255) << 2;
      bf16_t* d = sB + (size_t)n4 * LDB + (kp << 1);
      v2bf p0 = {(bf16_t)ca.x, (bf16_t)cb.x};
      v2bf p1 = {(bf16_t)ca.y, (bf16_t)cb.y};
      v2bf p2 = {(bf16_t)ca.z, (bf16_t)cb.z};
      v2bf p3 = {(bf16_t)ca.w, (bf16_t)cb.w};
      *reinterpret_cast<v2bf*>(d)           = p0;
      *reinterpret_cast<v2bf*>(d + LDB)     = p1;
      *reinterpret_cast<v2bf*>(d + 2 * LDB) = p2;
      *reinterpret_cast<v2bf*>(d + 3 * LDB) = p3;
    }
    ca = na; cb = nb;
  }
}

// ---- contiguous 81920B chunk copy: global(bf16 image) -> LDS ---------------

#if __has_builtin(__builtin_amdgcn_global_load_async_to_lds_b128)
#define USE_ASYNC_STAGE 1
#endif

#ifdef USE_ASYNC_STAGE
__device__ __forceinline__ void async_stage_wait() {
#if __has_builtin(__builtin_amdgcn_s_wait_asynccnt)
  __builtin_amdgcn_s_wait_asynccnt(0);
#else
  asm volatile("s_wait_asynccnt 0x0" ::: "memory");
#endif
}
// Fire-and-forget DMA: 10 x 16B per thread, no VGPR data, tracked by ASYNCcnt.
__device__ __forceinline__ void stage_copy(const bf16_t* __restrict__ src,
                                           bf16_t* dst, int tid) {
#pragma unroll
  for (int i = 0; i < 10; ++i) {
    size_t off = (size_t)(tid + i * NTHREADS) * 8;
    __builtin_amdgcn_global_load_async_to_lds_b128(
        (v4i*)(src + off), (v4i*)(dst + off), /*offset=*/0, /*cpol=*/0);
  }
}
#else
__device__ __forceinline__ void async_stage_wait() {}
// Synchronous fallback, software-pipelined (2 loads ahead).
__device__ __forceinline__ void stage_copy(const bf16_t* __restrict__ src,
                                           bf16_t* dst, int tid) {
  v8bf a, b, c;
  a = *reinterpret_cast<const v8bf*>(src + (size_t)tid * 8);
  b = *reinterpret_cast<const v8bf*>(src + (size_t)(tid + NTHREADS) * 8);
#pragma unroll
  for (int i = 0; i < 10; ++i) {
    if (i + 2 < 10)
      c = *reinterpret_cast<const v8bf*>(src + (size_t)(tid + (i + 2) * NTHREADS) * 8);
    *reinterpret_cast<v8bf*>(dst + (size_t)(tid + i * NTHREADS) * 8) = a;
    a = b; b = c;
  }
}
#endif

// ---- one-time converters ---------------------------------------------------

// One block per weight K-chunk: transpose/convert via LDS, dump contiguously.
__global__ __launch_bounds__(NTHREADS)
void convert_weights_kernel(const float* __restrict__ W1,
                            const float* __restrict__ W2,
                            bf16_t* __restrict__ wsW) {
  extern __shared__ char smem[];
  bf16_t* sB = reinterpret_cast<bf16_t*>(smem);
  const int tid = threadIdx.x;
  const int bid = blockIdx.x;
  const float* Wl;
  int kc;
  if (bid < WS_W2OFF) {                      // W1 chunk
    int e = bid / NCHUNK1, c = bid % NCHUNK1;
    Wl = W1 + (size_t)e * D_IN_K * H_N;
    kc = c * 32;
  } else {                                   // W2 chunk
    int b2 = bid - WS_W2OFF;
    int e = b2 / NCHUNK2, c = b2 % NCHUNK2;
    Wl = W2 + (size_t)e * H_N * H_N;
    kc = c * 32;
  }
  stage_w_chunk(Wl, kc, sB, tid);
  __syncthreads();
  bf16_t* dst = wsW + (size_t)bid * SB_ELEMS;
#pragma unroll
  for (int i = 0; i < 10; ++i) {
    size_t off = (size_t)(tid + i * NTHREADS) * 8;
    *reinterpret_cast<v8bf*>(dst + off) = *reinterpret_cast<const v8bf*>(sB + off);
  }
}

__global__ __launch_bounds__(NTHREADS)
void convert_x_kernel(const float* __restrict__ x, bf16_t* __restrict__ xb) {
  size_t i = ((size_t)blockIdx.x * NTHREADS + threadIdx.x) * 8;
  float4 a = *reinterpret_cast<const float4*>(x + i);
  float4 b = *reinterpret_cast<const float4*>(x + i + 4);
  v8bf p = {(bf16_t)a.x, (bf16_t)a.y, (bf16_t)a.z, (bf16_t)a.w,
            (bf16_t)b.x, (bf16_t)b.y, (bf16_t)b.z, (bf16_t)b.w};
  *reinterpret_cast<v8bf*>(xb + i) = p;
}

// ---- fused main kernel -----------------------------------------------------

__global__ __launch_bounds__(NTHREADS, 1)
void ensemble_critic_fused(const bf16_t* __restrict__ xb,
                           const bf16_t* __restrict__ wsW,
                           const float* __restrict__ b1,
                           const float* __restrict__ g1, const float* __restrict__ be1,
                           const float* __restrict__ b2,
                           const float* __restrict__ g2, const float* __restrict__ be2,
                           const float* __restrict__ W3, const float* __restrict__ b3,
                           float* __restrict__ out)
{
  extern __shared__ char smem[];
  // [h1t 132096][sB0 81920][sB1 81920][sA0 5120][sA1 5120][red 768] = 306944 B
  bf16_t* h1t    = reinterpret_cast<bf16_t*>(smem);
  bf16_t* sBbase = reinterpret_cast<bf16_t*>(smem + 132096);
  bf16_t* sAbase = reinterpret_cast<bf16_t*>(smem + 132096 + 2 * 81920);
  float*  redS   = reinterpret_cast<float*>(smem + 132096 + 2 * 81920 + 2 * 5120);
  float*  redQ   = redS + BT;
  float*  qrow   = redQ + BT;

  const int tid  = threadIdx.x;
  const int lane = tid & 31;
  const int wid  = tid >> 5;          // 0..15
  const int l15  = lane & 15;
  const int hi   = lane >> 4;         // 0 / 1
  const int mb   = wid >> 2;          // wave's M block (0..3), 16 rows
  const int nb   = wid & 3;           // wave's N block (0..3), 256 cols
  const int ncBase = nb * 256;
  const int mLoc   = mb * 16 + hi * 8;   // C-reg j -> local row (mLoc + j)
  const int rBase  = blockIdx.x * BT;
  const int e      = blockIdx.y;
  const size_t eH  = (size_t)e * H_N;

  const v8f vzero = {0.f,0.f,0.f,0.f,0.f,0.f,0.f,0.f};
  v8f acc[16];
#pragma unroll
  for (int nt = 0; nt < 16; ++nt) acc[nt] = vzero;

  // ----------------------------------------------------------------- GEMM 1
  {
    const bf16_t* Wc = wsW + (size_t)(e * NCHUNK1) * SB_ELEMS;
    stage_copy(Wc, sBbase, tid);
    {
      int r = tid >> 3, kq = (tid & 7) << 2;
      v4bf v = *reinterpret_cast<const v4bf*>(xb + (size_t)(rBase + r) * D_IN_K + kq);
      *reinterpret_cast<v4bf*>(sAbase + r * LDB + kq) = v;
    }
    for (int it = 0; it < NCHUNK1; ++it) {
      async_stage_wait();                     // async DMA of cur complete
      __syncthreads();                        // cur staged; prev compute done
      bf16_t* sBcur = sBbase + (it & 1) * SB_ELEMS;
      bf16_t* sBnxt = sBbase + ((it + 1) & 1) * SB_ELEMS;
      bf16_t* sAcur = sAbase + (it & 1) * SA_ELEMS;
      bf16_t* sAnxt = sAbase + ((it + 1) & 1) * SA_ELEMS;
      if (it + 1 < NCHUNK1) {
        stage_copy(Wc + (size_t)(it + 1) * SB_ELEMS, sBnxt, tid);
        int r = tid >> 3, kq = (tid & 7) << 2;
        v4bf v = *reinterpret_cast<const v4bf*>(
            xb + (size_t)(rBase + r) * D_IN_K + (it + 1) * 32 + kq);
        *reinterpret_cast<v4bf*>(sAnxt + r * LDB + kq) = v;
      }
      v16bf afr = load_frag(sAcur + (mb * 16 + l15) * LDB, hi);
#pragma unroll
      for (int nt = 0; nt < 16; ++nt) {
        v16bf bfr = load_frag(sBcur + (ncBase + nt * 16 + l15) * LDB, hi);
        acc[nt] = __builtin_amdgcn_wmma_f32_16x16x32_bf16(
            false, afr, false, bfr, (short)0, acc[nt], false, false);
      }
    }
  }

  float mean[8], rstd[8];

  // -------------------------------------------- bias1 + LayerNorm1 + ReLU
#pragma unroll
  for (int nt = 0; nt < 16; ++nt) {
    float bv = b1[eH + ncBase + nt * 16 + l15];
#pragma unroll
    for (int j = 0; j < 8; ++j) acc[nt][j] += bv;
  }
  __syncthreads();                            // last GEMM1 compute done
  if (tid < BT) { redS[tid] = 0.f; redQ[tid] = 0.f; }
  __syncthreads();
#pragma unroll
  for (int j = 0; j < 8; ++j) {
    float s = 0.f, q = 0.f;
#pragma unroll
    for (int nt = 0; nt < 16; ++nt) { float v = acc[nt][j]; s += v; q += v * v; }
#pragma unroll
    for (int m = 1; m < 16; m <<= 1) { s += __shfl_xor(s, m, 32); q += __shfl_xor(q, m, 32); }
    if (l15 == 0) { atomicAdd(&redS[mLoc + j], s); atomicAdd(&redQ[mLoc + j], q); }
  }
  __syncthreads();
#pragma unroll
  for (int j = 0; j < 8; ++j) {
    float mu = redS[mLoc + j] * (1.f / H_N);
    float va = redQ[mLoc + j] * (1.f / H_N) - mu * mu;
    mean[j] = mu;
    rstd[j] = rsqrtf(va + LN_EPS);
  }
#pragma unroll
  for (int nt = 0; nt < 16; ++nt) {
    int col = ncBase + nt * 16 + l15;
    float g  = g1[eH + col];
    float bb = be1[eH + col];
#pragma unroll
    for (int j = 0; j < 8; ++j) {
      float v = (acc[nt][j] - mean[j]) * rstd[j] * g + bb;
      v = fmaxf(v, 0.f);
      h1t[(mLoc + j) * LDH + col] = (bf16_t)v;
    }
  }
  __syncthreads();                            // h1t ready; sB buffers free

  // ----------------------------------------------------------------- GEMM 2
#pragma unroll
  for (int nt = 0; nt < 16; ++nt) acc[nt] = vzero;
  {
    const bf16_t* Wc = wsW + (size_t)(WS_W2OFF + e * NCHUNK2) * SB_ELEMS;
    stage_copy(Wc, sBbase, tid);
    for (int it = 0; it < NCHUNK2; ++it) {
      async_stage_wait();
      __syncthreads();
      bf16_t* sBcur = sBbase + (it & 1) * SB_ELEMS;
      bf16_t* sBnxt = sBbase + ((it + 1) & 1) * SB_ELEMS;
      if (it + 1 < NCHUNK2) stage_copy(Wc + (size_t)(it + 1) * SB_ELEMS, sBnxt, tid);
      v16bf afr = load_frag(h1t + (mb * 16 + l15) * LDH + it * 32, hi);
#pragma unroll
      for (int nt = 0; nt < 16; ++nt) {
        v16bf bfr = load_frag(sBcur + (ncBase + nt * 16 + l15) * LDB, hi);
        acc[nt] = __builtin_amdgcn_wmma_f32_16x16x32_bf16(
            false, afr, false, bfr, (short)0, acc[nt], false, false);
      }
    }
  }

  // ----------------- bias2 + LayerNorm2 + ReLU + head dot + qs write
#pragma unroll
  for (int nt = 0; nt < 16; ++nt) {
    float bv = b2[eH + ncBase + nt * 16 + l15];
#pragma unroll
    for (int j = 0; j < 8; ++j) acc[nt][j] += bv;
  }
  __syncthreads();                            // last GEMM2 compute done
  if (tid < BT) { redS[tid] = 0.f; redQ[tid] = 0.f; qrow[tid] = 0.f; }
  __syncthreads();
#pragma unroll
  for (int j = 0; j < 8; ++j) {
    float s = 0.f, q = 0.f;
#pragma unroll
    for (int nt = 0; nt < 16; ++nt) { float v = acc[nt][j]; s += v; q += v * v; }
#pragma unroll
    for (int m = 1; m < 16; m <<= 1) { s += __shfl_xor(s, m, 32); q += __shfl_xor(q, m, 32); }
    if (l15 == 0) { atomicAdd(&redS[mLoc + j], s); atomicAdd(&redQ[mLoc + j], q); }
  }
  __syncthreads();
#pragma unroll
  for (int j = 0; j < 8; ++j) {
    float mu = redS[mLoc + j] * (1.f / H_N);
    float va = redQ[mLoc + j] * (1.f / H_N) - mu * mu;
    mean[j] = mu;
    rstd[j] = rsqrtf(va + LN_EPS);
  }
  float hacc[8] = {0.f,0.f,0.f,0.f,0.f,0.f,0.f,0.f};
#pragma unroll
  for (int nt = 0; nt < 16; ++nt) {
    int col = ncBase + nt * 16 + l15;
    float g   = g2[eH + col];
    float bb  = be2[eH + col];
    float w3v = W3[eH + col];
#pragma unroll
    for (int j = 0; j < 8; ++j) {
      float v = (acc[nt][j] - mean[j]) * rstd[j] * g + bb;
      v = fmaxf(v, 0.f);
      hacc[j] += v * w3v;
    }
  }
#pragma unroll
  for (int j = 0; j < 8; ++j) {
    float s = hacc[j];
#pragma unroll
    for (int m = 1; m < 16; m <<= 1) s += __shfl_xor(s, m, 32);
    if (l15 == 0) atomicAdd(&qrow[mLoc + j], s);
  }
  __syncthreads();
  if (tid < BT) {
    // out layout: [ q (B) | qs (E*B) ]
    out[B_TOT + (size_t)e * B_TOT + rBase + tid] = qrow[tid] + b3[e];
  }
}

// q[b] = min_e qs[e][b]
__global__ void min_reduce_kernel(float* __restrict__ out) {
  int b = blockIdx.x * blockDim.x + threadIdx.x;
  float m = out[B_TOT + b];
#pragma unroll
  for (int e = 1; e < E_N; ++e) m = fminf(m, out[B_TOT + (size_t)e * B_TOT + b]);
  out[b] = m;
}

extern "C" void kernel_launch(void* const* d_in, const int* in_sizes, int n_in,
                              void* d_out, int out_size, void* d_ws, size_t ws_size,
                              hipStream_t stream) {
  (void)in_sizes; (void)n_in; (void)out_size; (void)ws_size;
  const float* x   = (const float*)d_in[0];
  const float* W1  = (const float*)d_in[1];
  const float* b1  = (const float*)d_in[2];
  const float* g1  = (const float*)d_in[3];
  const float* be1 = (const float*)d_in[4];
  const float* W2  = (const float*)d_in[5];
  const float* b2  = (const float*)d_in[6];
  const float* g2  = (const float*)d_in[7];
  const float* be2 = (const float*)d_in[8];
  const float* W3  = (const float*)d_in[9];
  const float* b3  = (const float*)d_in[10];
  float* out = (float*)d_out;

  bf16_t* wsW = (bf16_t*)d_ws;   // 384 chunk images (31.5 MB)
  bf16_t* xb  = wsW + (size_t)(E_N * (NCHUNK1 + NCHUNK2)) * SB_ELEMS; // x bf16 (16 MB)

  // Pass 1: one-time per-launch conversions
  convert_weights_kernel<<<E_N * (NCHUNK1 + NCHUNK2), NTHREADS, 81920, stream>>>(W1, W2, wsW);
  convert_x_kernel<<<(B_TOT * D_IN_K) / (NTHREADS * 8), NTHREADS, 0, stream>>>(x, xb);

  // Pass 2: fused ensemble forward
  size_t smem = 132096            // h1 tile (bf16, 64 x 1032)
              + 2 * 81920         // double-buffered weight chunk (bf16, 1024 x 40)
              + 2 * 5120          // double-buffered A staging (bf16, 64 x 40)
              + 3 * BT * sizeof(float);
  dim3 grid(B_TOT / BT, E_N);
  ensemble_critic_fused<<<grid, NTHREADS, smem, stream>>>(
      xb, wsW, b1, g1, be1, b2, g2, be2, W3, b3, out);

  // Pass 3: min over ensemble
  min_reduce_kernel<<<B_TOT / 256, 256, 0, stream>>>(out);
}
